// TransformerEncoder_76020921140061
// MI455X (gfx1250) — compile-verified, matrix operand order checked by reading, not run on hardware
//
#include <hip/hip_runtime.h>
#include <hip/hip_bf16.h>

typedef __attribute__((ext_vector_type(16))) _Float16 v16h;
typedef __attribute__((ext_vector_type(8)))  _Float16 v8h;
typedef __attribute__((ext_vector_type(8)))  float    v8f;
typedef __attribute__((ext_vector_type(4)))  unsigned int u32x4;
typedef __attribute__((ext_vector_type(8)))  int      i32x8;
typedef __attribute__((ext_vector_type(4)))  int      i32x4;

#define DEV static __device__ __forceinline__

static constexpr int Bn = 4, Cc = 256, HW = 64 * 64, Mpix = Bn * HW; // 16384
static constexpr int NELEM = Bn * Cc * HW;                           // 4M
static constexpr int LDSP = 264;  // 256 f16 + 4-dword pad per row (bank-conflict free)
static constexpr unsigned WBUFB = 64 * LDSP * 2;  // bytes per LDS weight buffer

#if __has_builtin(__builtin_amdgcn_tensor_load_to_lds) && \
    __has_builtin(__builtin_amdgcn_s_wait_tensorcnt)
#define HAVE_TDM 1
#endif

// ---------------------------------------------------------------------------
// Fragment load: 16x32 f16 operand (A or B) from row-major-in-K memory.
// Lane L<16 holds K {0..7,16..23}; lane L>=16 holds K {8..15,24..31}.
// `row` must already include (lane & 15).
// ---------------------------------------------------------------------------
DEV v16h load_frag(const _Float16* base, int row, int stride, int koff) {
  const int hi = (threadIdx.x >> 4) & 1;
  const _Float16* p = base + (size_t)row * stride + koff + hi * 8;
  v8h lo = *(const v8h*)(p);
  v8h hh = *(const v8h*)(p + 16);
  v16h r;
#pragma unroll
  for (int i = 0; i < 8; ++i) { r[i] = lo[i]; r[i + 8] = hh[i]; }
  return r;
}

DEV v8f wmma_f16(v16h a, v16h b, v8f c) {
  return __builtin_amdgcn_wmma_f32_16x16x32_f16(false, a, false, b, (short)0, c,
                                                false, false);
}

DEV float lrelu(float v) { return v >= 0.f ? v : 0.2f * v; }

// Compiler-level escape: TDM writes LDS behind the optimizer's back; make the
// shared buffer "potentially modified" so ds reads are not folded to undef.
DEV void lds_written_fence(const _Float16* sw) {
  asm volatile("" : : "r"(sw) : "memory");
}

// ---------------------------------------------------------------------------
// TDM: DMA a [rows x 256] f16 tile (global row stride `strideElems`) into LDS
// at byte offset `ldsOff`, padding each 512B row with 16B (LDSP layout).
// Issued by one wave; caller must s_wait_tensorcnt + barrier.
// ---------------------------------------------------------------------------
#ifdef HAVE_TDM
DEV void tdm_load_w(const _Float16* g, unsigned ldsOff, int rows, int strideElems) {
  unsigned long long ga = (unsigned long long)(uintptr_t)g;
  u32x4 g0 = {};
  g0[0] = 1u;                                       // count=1, user descriptor
  g0[1] = ldsOff;                                   // lds_addr (bytes)
  g0[2] = (unsigned)ga;                             // global_addr[31:0]
  g0[3] = (unsigned)((ga >> 32) & 0x1FFFFFFu) | (2u << 30);  // addr[56:32]|type=2
  i32x8 g1 = {};
  // data_size=2B, pad_enable, pad_interval=128dw (code 6), pad_amount=4dw (code 3)
  g1[0] = (int)((1u << 16) | (1u << 20) | (6u << 22) | (3u << 25));
  g1[1] = (int)(256u << 16);                        // tensor_dim0 = 256 (lo16)
  g1[2] = 0;                                        // td0 hi | td1 lo (td1=1<<20)
  g1[3] = (int)(16u | (256u << 16));                // td1 hi=16, tile_dim0=256
  g1[4] = rows & 0xffff;                            // tile_dim1=rows, tile_dim2=0
  g1[5] = strideElems;                              // tensor_dim0_stride (lo32)
  g1[6] = 0; g1[7] = 0;
  i32x4 g2 = {}; i32x4 g3 = {};
#if __clang_major__ >= 23
  i32x8 g4 = {};
  __builtin_amdgcn_tensor_load_to_lds(g0, g1, g2, g3, g4, 0);
#else
  __builtin_amdgcn_tensor_load_to_lds(g0, g1, g2, g3, 0);
#endif
}
#endif

// Fallback cooperative copy into padded LDS (v8h granularity).
DEV void coop_load_w(_Float16* sw, const _Float16* g, int rows, int strideElems) {
  for (int idx = threadIdx.x; idx < rows * 32; idx += blockDim.x) {
    int row = idx >> 5;
    int col = (idx & 31) * 8;
    *(v8h*)&sw[row * LDSP + col] = *(const v8h*)&g[(size_t)row * strideElems + col];
  }
}

// Stage a weight tile into LDS (block-cooperative, incl. sync).
DEV void stage_w(_Float16* sw, const _Float16* g, int rows, int strideElems) {
#ifdef HAVE_TDM
  if ((threadIdx.x >> 5) == 0) {
    tdm_load_w(g, 0u, rows, strideElems);
    __builtin_amdgcn_s_wait_tensorcnt(0);
  }
  lds_written_fence(sw);
#else
  coop_load_w(sw, g, rows, strideElems);
#endif
  __syncthreads();
}

// ---------------------------------------------------------------------------
// Elementwise prep kernels
// ---------------------------------------------------------------------------
__global__ void k_x_to_pm(const float* __restrict__ x, _Float16* __restrict__ xp) {
  int idx = blockIdx.x * 256 + threadIdx.x;
  if (idx >= NELEM) return;
  int xq = idx & 63; int t = idx >> 6;
  int y = t & 63; t >>= 6;
  int c = t & 255; int b = t >> 8;
  int pix = b * HW + y * 64 + xq;
  xp[(size_t)pix * Cc + c] = (_Float16)x[idx];
}

__global__ void k_cvt_f16(const float* __restrict__ src, _Float16* __restrict__ dst, int n) {
  int i = blockIdx.x * 256 + threadIdx.x;
  if (i < n) dst[i] = (_Float16)src[i];
}

// w: [co][ci][3][3]  ->  wt: [co][tap][ci]
__global__ void k_w3x3_prep(const float* __restrict__ w, _Float16* __restrict__ wt) {
  int i = blockIdx.x * 256 + threadIdx.x;
  if (i >= 256 * 9 * 256) return;
  int ci = i & 255; int tap = (i >> 8) % 9; int co = i / (9 * 256);
  wt[i] = (_Float16)w[((co * 256 + ci) * 9) + tap];
}

// ---------------------------------------------------------------------------
// 1x1 conv GEMM, 4 waves/block: 64 pixels x 64 co. Weights TDM-staged in LDS.
// ---------------------------------------------------------------------------
__global__ __launch_bounds__(128) void k_gemm_qkv(
    const _Float16* __restrict__ A, const _Float16* __restrict__ W,
    const float* __restrict__ bias, _Float16* __restrict__ out) {
  __shared__ _Float16 sw[64 * LDSP];
  const int lane = threadIdx.x & 31;
  const int wave = threadIdx.x >> 5;
  const int m0 = blockIdx.x * 64 + wave * 16;
  const int n0 = blockIdx.y * 64;

  stage_w(sw, W + (size_t)n0 * 256, 64, 256);

  const int arow = m0 + (lane & 15);
  v16h afr[8];
#pragma unroll
  for (int kc = 0; kc < 8; ++kc) afr[kc] = load_frag(A, arow, 256, kc * 32);

  v8f z = {};
  v8f acc[4] = {z, z, z, z};
#pragma unroll
  for (int kc = 0; kc < 8; ++kc) {
#pragma unroll
    for (int j = 0; j < 4; ++j) {
      v16h bf = load_frag(sw, 16 * j + (lane & 15), LDSP, kc * 32);
      acc[j] = wmma_f16(afr[kc], bf, acc[j]);
    }
  }
  const int rbase = ((lane >> 4) & 1) * 8;
#pragma unroll
  for (int j = 0; j < 4; ++j) {
    int col = n0 + 16 * j + (lane & 15);
    float bv = bias[col];
#pragma unroll
    for (int vi = 0; vi < 8; ++vi) {
      int row = m0 + rbase + vi;
      out[(size_t)row * 256 + col] = (_Float16)(acc[j][vi] + bv);
    }
  }
}

// ---------------------------------------------------------------------------
// Implicit-GEMM 3x3 conv, 4 waves/block: 64 pixels (one image row) x 64 co.
// Per tap: weight slice [64co x 256ci] staged to LDS. With TDM, the DMA for
// tap t+1 is issued into the other LDS buffer before computing tap t
// (in-order tensor ops + s_wait_tensorcnt(1) => tap t resident).
// EPI==1: x1 = lrelu(conv+b)+residual, f32+f16 out.  EPI==0: raw f32 out.
// ---------------------------------------------------------------------------
template <int DIL, int EPI>
__global__ __launch_bounds__(128) void k_conv3(
    const _Float16* __restrict__ in, const _Float16* __restrict__ wt,
    const float* __restrict__ bias, const float* __restrict__ xres,
    float* __restrict__ outf, _Float16* __restrict__ outh) {
  __shared__ _Float16 sw[2][64 * LDSP];
  const int lane = threadIdx.x & 31;
  const int wave = threadIdx.x >> 5;
  const int pt = blockIdx.x * 64 + wave * 16;   // block = one full image row
  const int n0 = blockIdx.y * 64;
  const int b = pt / HW; const int rem = pt % HW;
  const int y = rem / 64; const int x0 = rem % 64;
  const int mx = x0 + (lane & 15);
  const _Float16* wbase = wt + (size_t)n0 * (9 * 256);
  v8f z = {};
  v8f acc[4] = {z, z, z, z};

#ifdef HAVE_TDM
  if (wave == 0) tdm_load_w(wbase, 0u, 64, 9 * 256);  // prologue: tap 0 -> buf 0
#endif
  for (int tap = 0; tap < 9; ++tap) {
    __syncthreads();  // prior readers of the buffer being (over)written are done
#ifdef HAVE_TDM
    if (wave == 0) {
      if (tap < 8) {
        tdm_load_w(wbase + (tap + 1) * 256, ((tap + 1) & 1) * WBUFB, 64, 9 * 256);
        __builtin_amdgcn_s_wait_tensorcnt(1);  // tap resident, tap+1 in flight
      } else {
        __builtin_amdgcn_s_wait_tensorcnt(0);
      }
    }
    lds_written_fence(&sw[0][0]);
    __syncthreads();
    const _Float16* wcur = &sw[tap & 1][0];
#else
    coop_load_w(&sw[0][0], wbase + tap * 256, 64, 9 * 256);
    __syncthreads();
    const _Float16* wcur = &sw[0][0];
#endif

    const int ky = tap / 3, kx = tap % 3;
    int yy = y + (ky - 1) * DIL;
    int xx = mx + (kx - 1) * DIL;
    bool ok = ((unsigned)yy < 64u) && ((unsigned)xx < 64u);
    int apix = b * HW + yy * 64 + xx;

    v16h afr[8];
#pragma unroll
    for (int kc = 0; kc < 8; ++kc) {
      v16h a = {};
      if (ok) a = load_frag(in, apix, 256, kc * 32);  // per-lane predicated
      afr[kc] = a;
    }
#pragma unroll
    for (int kc = 0; kc < 8; ++kc) {
#pragma unroll
      for (int j = 0; j < 4; ++j) {
        v16h bf = load_frag(wcur, 16 * j + (lane & 15), LDSP, kc * 32);
        acc[j] = wmma_f16(afr[kc], bf, acc[j]);       // full EXEC here
      }
    }
  }

  const int rbase = ((lane >> 4) & 1) * 8;
#pragma unroll
  for (int j = 0; j < 4; ++j) {
    int col = n0 + 16 * j + (lane & 15);
    float bv = bias[col];
#pragma unroll
    for (int vi = 0; vi < 8; ++vi) {
      int px = x0 + rbase + vi;
      int pix = b * HW + y * 64 + px;
      float v = acc[j][vi] + bv;
      if (EPI == 1) {
        float r = xres[(((size_t)b * 256 + col) * 64 + y) * 64 + px];
        v = lrelu(v) + r;
        outf[(size_t)pix * 256 + col] = v;
        outh[(size_t)pix * 256 + col] = (_Float16)v;
      } else {
        outf[(size_t)pix * 256 + col] = v;
      }
    }
  }
}

// ---------------------------------------------------------------------------
// Patchify one branch: q,k -> [problem][N][dh] row-major; v -> [problem][dh][N]
// problem = b*heads + head; heads = s; dh = 64*s; N = (64/s)^2. 2^20 elem each.
// ---------------------------------------------------------------------------
__global__ void k_patchify(const _Float16* __restrict__ qp,
                           const _Float16* __restrict__ kp,
                           const _Float16* __restrict__ vp,
                           _Float16* __restrict__ Qb, _Float16* __restrict__ Kb,
                           _Float16* __restrict__ Vtb, int s, int branch) {
  const int dc = 64, hn = s, nw = 64 / s;
  const int N = nw * nw, dh = dc * s, cph = dc / s;
  int idx = blockIdx.x * 256 + threadIdx.x;
  if (idx >= (1 << 20)) return;
  int d = idx % dh;
  int n = (idx / dh) % N;
  int p = idx / (dh * N);
  int b = p / hn, head = p % hn;
  int c_local = d / (s * s); int r = d % (s * s);
  int dy = r / s, dx = r % s;
  int iy = n / nw, ix = n % nw;
  int pix = b * HW + (iy * s + dy) * 64 + (ix * s + dx);
  int ch = branch * dc + head * cph + c_local;
  size_t src = (size_t)pix * 256 + ch;
  Qb[idx] = qp[src];
  Kb[idx] = kp[src];
  Vtb[(size_t)p * (N * dh) + (size_t)d * N + n] = vp[src];
}

// ---------------------------------------------------------------------------
// Flash attention. One wave = 16 queries x DHS output dims of one problem.
// S^T = K.Q^T so softmax-over-keys is per-lane + one shfl_xor(16); the two
// stacked S^T C-tiles map directly onto the P B-fragment (no shuffles).
// grid = (N/16, DH/DHS, problems)
// ---------------------------------------------------------------------------
template <int S, int N, int DH, int DHS>
__global__ __launch_bounds__(32) void k_attn(
    const _Float16* __restrict__ Qb, const _Float16* __restrict__ Kb,
    const _Float16* __restrict__ Vtb, _Float16* __restrict__ att, int branch) {
  const int lane = threadIdx.x & 31;
  const int q0 = blockIdx.x * 16;
  const int slice = blockIdx.y;
  const int p = blockIdx.z;
  const _Float16* Q  = Qb  + (size_t)p * N * DH;
  const _Float16* K  = Kb  + (size_t)p * N * DH;
  const _Float16* Vt = Vtb + (size_t)p * N * DH;
  const int qrow = q0 + (lane & 15);
  const int d0 = slice * DHS;

  v16h qf[DH / 32];
#pragma unroll
  for (int kk = 0; kk < DH / 32; ++kk) qf[kk] = load_frag(Q, qrow, DH, kk * 32);

  const float scale = 1.0f / __fsqrt_rn((float)DH);
  float mrun = -1e30f, lrun = 0.f;
  v8f z = {};
  v8f o[DHS / 16];
#pragma unroll
  for (int t = 0; t < DHS / 16; ++t) o[t] = z;

  for (int kb = 0; kb < N; kb += 32) {
    v8f s0 = z, s1 = z;
#pragma unroll
    for (int kk = 0; kk < DH / 32; ++kk) {
      v16h a0 = load_frag(K, kb + (lane & 15), DH, kk * 32);
      v16h a1 = load_frag(K, kb + 16 + (lane & 15), DH, kk * 32);
      s0 = wmma_f16(a0, qf[kk], s0);
      s1 = wmma_f16(a1, qf[kk], s1);
    }
    float bm = -1e30f;
#pragma unroll
    for (int i = 0; i < 8; ++i) {
      s0[i] *= scale; s1[i] *= scale;
      bm = fmaxf(bm, fmaxf(s0[i], s1[i]));
    }
    bm = fmaxf(bm, __shfl_xor(bm, 16, 32));   // combine key halves (lane pair)
    float mnew = fmaxf(mrun, bm);
    float corr = __expf(mrun - mnew);
    float ps = 0.f;
    v16h pf;
#pragma unroll
    for (int i = 0; i < 8; ++i) {
      float p0 = __expf(s0[i] - mnew);
      float p1 = __expf(s1[i] - mnew);
      ps += p0 + p1;
      pf[i] = (_Float16)p0;       // S^T tile0 C-layout == first 8 B halves
      pf[i + 8] = (_Float16)p1;   // S^T tile1 C-layout == last 8 B halves
    }
    ps += __shfl_xor(ps, 16, 32);
    lrun = lrun * corr + ps;
    mrun = mnew;
#pragma unroll
    for (int t = 0; t < DHS / 16; ++t) {
#pragma unroll
      for (int i = 0; i < 8; ++i) o[t][i] *= corr;
      v16h av = load_frag(Vt, d0 + t * 16 + (lane & 15), N, kb);
      o[t] = wmma_f16(av, pf, o[t]);
    }
  }
  float inv = 1.0f / lrun;
  // un-patchify scatter: lane = query column, rows = dh dims
  const int nq = q0 + (lane & 15);
  const int dc = 64, cph = dc / S, nw = 64 / S, hn = S;
  const int b = p / hn, head = p % hn;
  const int iy = nq / nw, ix = nq % nw;
  const int rbase = ((lane >> 4) & 1) * 8;
#pragma unroll
  for (int t = 0; t < DHS / 16; ++t) {
#pragma unroll
    for (int vi = 0; vi < 8; ++vi) {
      int d = d0 + t * 16 + rbase + vi;
      int c_local = d / (S * S); int r = d % (S * S);
      int dy = r / S, dx = r % S;
      int pix = b * HW + (iy * S + dy) * 64 + (ix * S + dx);
      int ch = branch * dc + head * cph + c_local;
      att[(size_t)pix * 256 + ch] = (_Float16)(o[t][vi] * inv);
    }
  }
}

// ---------------------------------------------------------------------------
// Instance norm + leaky ReLU over pixels per (b, c). z f32 pixel-major in.
// ---------------------------------------------------------------------------
__global__ void k_instnorm(const float* __restrict__ zin, _Float16* __restrict__ yh,
                           float* __restrict__ yf) {
  __shared__ float ssum[256], ssq[256];
  int bc = blockIdx.x; int b = bc >> 8; int c = bc & 255;
  const float* base = zin + (size_t)b * HW * 256 + c;
  float s = 0.f, q = 0.f;
  for (int pi = threadIdx.x; pi < HW; pi += 256) {
    float v = base[(size_t)pi * 256];
    s += v; q += v * v;
  }
  ssum[threadIdx.x] = s; ssq[threadIdx.x] = q;
  __syncthreads();
  for (int off = 128; off > 0; off >>= 1) {
    if (threadIdx.x < off) {
      ssum[threadIdx.x] += ssum[threadIdx.x + off];
      ssq[threadIdx.x]  += ssq[threadIdx.x + off];
    }
    __syncthreads();
  }
  float mean = ssum[0] / (float)HW;
  float var  = ssq[0] / (float)HW - mean * mean;
  float rstd = __frsqrt_rn(var + 1e-5f);
  for (int pi = threadIdx.x; pi < HW; pi += 256) {
    float v = base[(size_t)pi * 256];
    float o = lrelu((v - mean) * rstd);
    size_t dst = ((size_t)b * HW + pi) * 256 + c;
    yh[dst] = (_Float16)o;
    if (yf) yf[dst] = o;
  }
}

__global__ void k_final(const float* __restrict__ x1f, const float* __restrict__ y2f,
                        float* __restrict__ out) {
  int idx = blockIdx.x * 256 + threadIdx.x;
  if (idx >= NELEM) return;
  int xq = idx & 63; int t = idx >> 6;
  int y = t & 63; t >>= 6;
  int c = t & 255; int b = t >> 8;
  size_t pc = ((size_t)(b * HW + y * 64 + xq)) * 256 + c;
  out[idx] = x1f[pc] + y2f[pc];
}

// ---------------------------------------------------------------------------
extern "C" void kernel_launch(void* const* d_in, const int* in_sizes, int n_in,
                              void* d_out, int out_size, void* d_ws, size_t ws_size,
                              hipStream_t stream) {
  (void)in_sizes; (void)n_in; (void)out_size; (void)ws_size;
  const float* x   = (const float*)d_in[0];
  const float* wq  = (const float*)d_in[2];  const float* bq  = (const float*)d_in[3];
  const float* wk  = (const float*)d_in[4];  const float* bk  = (const float*)d_in[5];
  const float* wv  = (const float*)d_in[6];  const float* bv  = (const float*)d_in[7];
  const float* wo  = (const float*)d_in[8];  const float* bo  = (const float*)d_in[9];
  const float* wf1 = (const float*)d_in[10]; const float* bf1 = (const float*)d_in[11];
  const float* wf2 = (const float*)d_in[12]; const float* bf2 = (const float*)d_in[13];

  char* ws = (char*)d_ws;
  size_t cur = 0;
  auto alloc = [&](size_t bytes) -> void* {
    void* ptr = ws + cur;
    cur += (bytes + 255) & ~(size_t)255;
    return ptr;
  };
  const size_t PM16 = (size_t)Mpix * 256 * sizeof(_Float16);  // 8 MB
  const size_t PM32 = (size_t)Mpix * 256 * sizeof(float);     // 16 MB
  const size_t BR16 = (size_t)4 * (1 << 20) * sizeof(_Float16);

  _Float16* xp   = (_Float16*)alloc(PM16);
  _Float16* qp   = (_Float16*)alloc(PM16);
  _Float16* kp   = (_Float16*)alloc(PM16);
  _Float16* vp   = (_Float16*)alloc(PM16);
  _Float16* Qb   = (_Float16*)alloc(BR16);
  _Float16* Kb   = (_Float16*)alloc(BR16);
  _Float16* Vtb  = (_Float16*)alloc(BR16);
  _Float16* attb = (_Float16*)alloc(PM16);
  float*    x1f  = (float*)alloc(PM32);
  _Float16* x1h  = (_Float16*)alloc(PM16);
  float*    zf   = (float*)alloc(PM32);   // reused for z1 then z2
  _Float16* y1h  = (_Float16*)alloc(PM16);
  float*    y2f  = (float*)alloc(PM32);
  _Float16* wqh  = (_Float16*)alloc(65536 * 2);
  _Float16* wkh  = (_Float16*)alloc(65536 * 2);
  _Float16* wvh  = (_Float16*)alloc(65536 * 2);
  _Float16* woh  = (_Float16*)alloc(589824 * 2);
  _Float16* wf1h = (_Float16*)alloc(589824 * 2);
  _Float16* wf2h = (_Float16*)alloc(589824 * 2);

  // ---- prep ----
  k_x_to_pm<<<(NELEM + 255) / 256, 256, 0, stream>>>(x, xp);
  k_cvt_f16<<<256, 256, 0, stream>>>(wq, wqh, 65536);
  k_cvt_f16<<<256, 256, 0, stream>>>(wk, wkh, 65536);
  k_cvt_f16<<<256, 256, 0, stream>>>(wv, wvh, 65536);
  k_w3x3_prep<<<2304, 256, 0, stream>>>(wo, woh);
  k_w3x3_prep<<<2304, 256, 0, stream>>>(wf1, wf1h);
  k_w3x3_prep<<<2304, 256, 0, stream>>>(wf2, wf2h);

  // ---- q, k, v 1x1-conv GEMMs (TDM weight staging) ----
  dim3 gg(Mpix / 64, 4);
  k_gemm_qkv<<<gg, 128, 0, stream>>>(xp, wqh, bq, qp);
  k_gemm_qkv<<<gg, 128, 0, stream>>>(xp, wkh, bk, kp);
  k_gemm_qkv<<<gg, 128, 0, stream>>>(xp, wvh, bv, vp);

  // ---- patchify all 4 branches (each exactly 2^20 elements) ----
  const int svals[4] = {1, 2, 4, 8};
  for (int i = 0; i < 4; ++i) {
    size_t boff = (size_t)i << 20;
    k_patchify<<<4096, 256, 0, stream>>>(qp, kp, vp, Qb + boff, Kb + boff,
                                         Vtb + boff, svals[i], i);
  }

  // ---- flash attention per branch ----
  k_attn<1, 4096,  64,  64><<<dim3(256, 1,  4), 32, 0, stream>>>(
      Qb,                 Kb,                 Vtb,                 attb, 0);
  k_attn<2, 1024, 128, 128><<<dim3( 64, 1,  8), 32, 0, stream>>>(
      Qb + ((size_t)1 << 20), Kb + ((size_t)1 << 20), Vtb + ((size_t)1 << 20), attb, 1);
  k_attn<4,  256, 256, 128><<<dim3( 16, 2, 16), 32, 0, stream>>>(
      Qb + ((size_t)2 << 20), Kb + ((size_t)2 << 20), Vtb + ((size_t)2 << 20), attb, 2);
  k_attn<8,   64, 512, 128><<<dim3(  4, 4, 32), 32, 0, stream>>>(
      Qb + ((size_t)3 << 20), Kb + ((size_t)3 << 20), Vtb + ((size_t)3 << 20), attb, 3);

  // ---- wo 3x3 conv + lrelu + residual -> x1 ----
  dim3 gc(Mpix / 64, 4);
  k_conv3<1, 1><<<gc, 128, 0, stream>>>(attb, woh, bo, x, x1f, x1h);

  // ---- feed-forward ----
  k_conv3<2, 0><<<gc, 128, 0, stream>>>(x1h, wf1h, bf1, nullptr, zf, nullptr);
  k_instnorm<<<Bn * 256, 256, 0, stream>>>(zf, y1h, nullptr);
  k_conv3<1, 0><<<gc, 128, 0, stream>>>(y1h, wf2h, bf2, nullptr, zf, nullptr);
  k_instnorm<<<Bn * 256, 256, 0, stream>>>(zf, y1h /*scratch*/, y2f);

  // ---- out = x1 + y (NCHW f32) ----
  k_final<<<(NELEM + 255) / 256, 256, 0, stream>>>(x1f, y2f, (float*)d_out);
}
